// MAD_4612794876395
// MI455X (gfx1250) — compile-verified, matrix operand order checked by reading, not run on hardware
//
#include <hip/hip_runtime.h>
#include <hip/hip_bf16.h>
#include <cstdint>

// Problem constants (from reference)
#define NDAT 100000
#define FIN  512      // IN_FEATS
#define HID_ 512      // HID
#define NC   128      // N_CLASSES
#define KNN  8        // N_NN
#define BSZ  2048     // BATCH
#define MROWS (BSZ + BSZ * KNN)   // 18432 rows: [date ; gathered train_dates]
#define NITER 8192                // K chunks of 32 in the field GEMM (f*16+h0/32)

typedef __attribute__((ext_vector_type(16))) __bf16 v16bf;
typedef __attribute__((ext_vector_type(8)))  float  v8f;
typedef __attribute__((ext_vector_type(4)))  float  v4f;   // native vec for NT loads

union Frag {
  v16bf v;
  uint4 q[2];
  __bf16 e[16];
};

// CDNA5 async global->LDS copy (16B per lane), tracked with ASYNCcnt.
__device__ __forceinline__ void async_copy_b128(uint32_t lds_off,
                                                const void* gptr) {
  asm volatile("global_load_async_to_lds_b128 %0, %1, off"
               :: "v"(lds_off), "v"(gptr)
               : "memory");
}
__device__ __forceinline__ void wait_async0() {
  asm volatile("s_wait_asynccnt 0" ::: "memory");
}

// Load a 16(rows)x32(k) WMMA operand fragment from a row-major [row][ldk] bf16
// array. Per ISA 7.12.2 (16-bit A 16x32): lanes 0-15 hold M=lane, K in
// {0..7,16..23}; lanes 16-31 hold M=lane-16, K in {8..15,24..31}.
// Two contiguous 16B LDS loads per lane; used for A and the n-major B operand.
__device__ __forceinline__ v16bf frag_load(const __bf16* base, int ldk,
                                           int row_base, int k0) {
  const int ln = threadIdx.x & 31;
  const int row = row_base + (ln & 15);
  const int kb = k0 + ((ln >> 4) << 3);
  const __bf16* p = base + row * ldk + kb;
  Frag f;
  f.q[0] = *(const uint4*)p;
  f.q[1] = *(const uint4*)(p + 16);
  return f.v;
}

// ---------------------------------------------------------------------------
// Kernel 0: one-time weight pre-swizzle (f32 -> bf16, fragment-ready layouts).
//  blocks [0,8192):  Wt[iter][c][hl] = field_w[f][(h0+hl)*NC+c],
//                    iter = f*16 + h0/32  (8KB bf16 tile per iter)
//  blocks [8192,8200): pos_wt[n][k] = pos_w[k][n]  (n-major bf16)
// field_w is streamed once from HBM with non-temporal loads.
// ---------------------------------------------------------------------------
__global__ __launch_bounds__(256) void wprep_kernel(
    const float* __restrict__ field_w, const float* __restrict__ pos_w,
    __bf16* __restrict__ Wt, __bf16* __restrict__ pos_wt) {
  __shared__ __align__(16) float tile[32 * NC];  // 16KB
  const int t = threadIdx.x;
  if (blockIdx.x < NITER) {
    const int iter = blockIdx.x;
    const int f = iter >> 4;
    const int h0 = (iter & 15) << 5;
    const float* src = field_w + (size_t)f * (HID_ * NC) + (size_t)h0 * NC;
    for (int e = t; e < 32 * 32; e += 256) {  // 1024 float4 granules
      const int i4 = e * 4;  // = hl*128 + c
      v4f v = __builtin_nontemporal_load((const v4f*)(src + i4));
      *(v4f*)(tile + i4) = v;
    }
    __syncthreads();
    // out element range [t*16, t*16+16): c = t>>1, hl = (t&1)*16 ..
    const int c = t >> 1, hl0 = (t & 1) << 4;
    Frag o;
    for (int i = 0; i < 16; ++i) o.e[i] = (__bf16)tile[(hl0 + i) * NC + c];
    uint4* dst = (uint4*)(Wt + (size_t)iter * 4096 + c * 32 + hl0);
    dst[0] = o.q[0];
    dst[1] = o.q[1];
  } else {
    const int bb = blockIdx.x - NITER;     // 0..7
    const int n = bb * 64 + (t >> 2);      // 512 rows total
    const int k0 = (t & 3) * 128;
    union { __bf16 e[8]; uint4 q; } u;
    for (int kk = 0; kk < 128; kk += 8) {
      for (int i = 0; i < 8; ++i)
        u.e[i] = (__bf16)pos_w[(size_t)(k0 + kk + i) * HID_ + n];
      *(uint4*)(pos_wt + (size_t)n * FIN + k0 + kk) = u.q;
    }
  }
}

// ---------------------------------------------------------------------------
// Kernel 1: pos_all[g][h] = G[g] @ pos_w + pos_b, G = [date ; train_dates[refs]]
// M=18432, N=512, K=512 bf16 WMMA GEMM. 576 blocks x 256 thr, Mtile=32.
// B slabs async-copied from pre-swizzled pos_wt.
// ---------------------------------------------------------------------------
__global__ __launch_bounds__(256) void pos_gemm_kernel(
    const int* __restrict__ idx, const float* __restrict__ date,
    const float* __restrict__ train_dates, const int* __restrict__ train_nns,
    const __bf16* __restrict__ pos_wt, const float* __restrict__ pos_b,
    float* __restrict__ pos_all) {
  __shared__ __align__(16) __bf16 As[32 * FIN];   // 32 M-rows x 512 k (32KB)
  __shared__ __align__(16) __bf16 Bt[HID_ * 32];  // 512 n-rows x 32 k (32KB)

  const int tid = threadIdx.x;
  const int g0 = blockIdx.x * 32;
  const uint32_t bt_base = (uint32_t)(uintptr_t)Bt;

  // Stage A: gather rows, f32 -> bf16. 8 threads per row, 64 floats each.
  {
    const int r = tid >> 3;
    const int sub = tid & 7;
    const int g = g0 + r;
    const float* src;
    if (g < BSZ) {
      src = date + (size_t)g * FIN;
    } else {
      const int p = g - BSZ;
      const int b = p >> 3, k = p & 7;
      const int ref = train_nns[idx[b] * KNN + k];
      src = train_dates + (size_t)ref * FIN;
    }
    const int c0 = sub * 64;
    for (int i = 0; i < 64; i += 4) {
      float4 v = *(const float4*)(src + c0 + i);
      __bf16* d = As + r * FIN + c0 + i;
      d[0] = (__bf16)v.x; d[1] = (__bf16)v.y;
      d[2] = (__bf16)v.z; d[3] = (__bf16)v.w;
    }
  }

  const int w = tid >> 5;  // wave 0..7, owns n-range [w*64, w*64+64)
  v8f acc[4][2] = {};

  for (int s = 0; s < 16; ++s) {  // 16 k-slabs of 32
    __syncthreads();
    // Stage Bt slab: pos_wt[n][s*32..+32] -> Bt[n][0..32], async 16B/lane.
    for (int j = 0; j < 8; ++j) {
      const int e = tid + j * 256;  // 2048 x 16B transfers
      async_copy_b128(bt_base + e * 16,
                      pos_wt + (size_t)(e >> 2) * FIN + s * 32 + (e & 3) * 8);
    }
    wait_async0();
    __syncthreads();
    const v16bf a0 = frag_load(As, FIN, 0, s * 32);
    const v16bf a1 = frag_load(As, FIN, 16, s * 32);
    for (int nt = 0; nt < 4; ++nt) {
      const v16bf bf = frag_load(Bt, 32, (w * 4 + nt) * 16, 0);
      acc[nt][0] = __builtin_amdgcn_wmma_f32_16x16x32_bf16(
          false, a0, false, bf, (short)0, acc[nt][0], false, false);
      acc[nt][1] = __builtin_amdgcn_wmma_f32_16x16x32_bf16(
          false, a1, false, bf, (short)0, acc[nt][1], false, false);
    }
  }

  // Epilogue: C layout — VGPR j: lanes 0-15 M=j, lanes 16-31 M=j+8.
  const int ln = tid & 31;
  const int mo = (ln >> 4) << 3;
  for (int nt = 0; nt < 4; ++nt) {
    const int n = (w * 4 + nt) * 16 + (ln & 15);
    const float pb = pos_b[n];
    for (int mf = 0; mf < 2; ++mf)
      for (int j = 0; j < 8; ++j) {
        const int g = g0 + mf * 16 + j + mo;
        pos_all[(size_t)g * HID_ + n] = acc[nt][mf][j] + pb;
      }
  }
}

// ---------------------------------------------------------------------------
// Kernel 2: per-b diff norms -> softmax dist -> wdiff, bf16 casts, and output
// initializer: out[b,c] = wmem@adapt_w + adapt_b + wdiff@field_b.
// 2048 blocks x 128 threads.
// ---------------------------------------------------------------------------
__global__ __launch_bounds__(128) void dist_kernel(
    const int* __restrict__ idx, const float* __restrict__ date,
    const float* __restrict__ mem, const int* __restrict__ train_nns,
    const float* __restrict__ adapt_w, const float* __restrict__ adapt_b,
    const float* __restrict__ field_b, const float* __restrict__ pos_all,
    __bf16* __restrict__ date_bf, __bf16* __restrict__ wdiff_bf,
    float* __restrict__ out) {
  __shared__ float red[KNN][128];
  __shared__ float dist_s[KNN];
  __shared__ float wdl[HID_];
  __shared__ float wmem_s[NC];
  __shared__ int refs_s[KNN];

  const int b = blockIdx.x, t = threadIdx.x;
  if (t < KNN) refs_s[t] = train_nns[idx[b] * KNN + t];
  __syncthreads();

  float pd[4], df[KNN][4];
  for (int i = 0; i < 4; ++i) pd[i] = pos_all[(size_t)b * HID_ + t + i * 128];
  for (int k = 0; k < KNN; ++k) {
    const float* pr = pos_all + (size_t)(BSZ + b * KNN + k) * HID_;
    float s = 0.f;
    for (int i = 0; i < 4; ++i) {
      const float d = pd[i] - pr[t + i * 128];
      df[k][i] = d;
      s += d * d;
    }
    red[k][t] = s;
  }
  __syncthreads();
  if (t < KNN) {  // one k per thread: serial 128-sum then norm
    float s = 0.f;
    for (int i = 0; i < 128; ++i) s += red[t][i];
    red[t][0] = sqrtf(s);
  }
  __syncthreads();
  if (t == 0) {  // softmax over -norm, K=8
    float mx = -1e30f;
    for (int k = 0; k < KNN; ++k) mx = fmaxf(mx, -red[k][0]);
    float e[KNN], sum = 0.f;
    for (int k = 0; k < KNN; ++k) { e[k] = __expf(-red[k][0] - mx); sum += e[k]; }
    for (int k = 0; k < KNN; ++k) dist_s[k] = e[k] / sum;
  }
  __syncthreads();
  for (int i = 0; i < 4; ++i) {
    const int h = t + i * 128;
    float wd = 0.f;
    for (int k = 0; k < KNN; ++k) wd += dist_s[k] * df[k][i];
    wdl[h] = wd;
    wdiff_bf[(size_t)b * HID_ + h] = (__bf16)wd;
    date_bf[(size_t)b * HID_ + h] = (__bf16)date[(size_t)b * HID_ + h];
  }
  {
    float wm = 0.f;
    for (int k = 0; k < KNN; ++k)
      wm += dist_s[k] * mem[(size_t)refs_s[k] * NC + t];
    wmem_s[t] = wm;
  }
  __syncthreads();
  float acc = adapt_b[t];  // sum_k dist == 1 (padded neighbor has zero weight)
  for (int c2 = 0; c2 < NC; ++c2) acc += wmem_s[c2] * adapt_w[c2 * NC + t];
  for (int h = 0; h < HID_; ++h) acc += wdl[h] * field_b[h * NC + t];
  out[(size_t)b * NC + t] = acc;
}

// ---------------------------------------------------------------------------
// Kernel 3: out_field[b,c] += sum_{f,h} date[b,f]*wdiff[b,h]*field_w[f,h,c]
// GEMM M=2048, N=128, K=262144; A synthesized on the fly (rank-1 rows).
// 256 blocks (64 M-tiles x 4 deterministic K-splits) x 256 thr, Mtile=32.
// B tiles are pre-swizzled bf16 (Wt) and async-copied, double-buffered.
// ---------------------------------------------------------------------------
__global__ __launch_bounds__(256) void field_gemm_kernel(
    const __bf16* __restrict__ Wt, const __bf16* __restrict__ date_bf,
    const __bf16* __restrict__ wdiff_bf, float* __restrict__ partial) {
  __shared__ __align__(16) __bf16 Bt[2][NC * 32];  // [c][k] per buffer, 8KB ea

  const int tid = threadIdx.x;
  const int mblk = blockIdx.x & 63;
  const int ks = blockIdx.x >> 6;  // K-split 0..3
  const int b0 = mblk * 32;
  const int iter0 = ks * (NITER / 4);
  const int w = tid >> 5, ln = tid & 31;
  const int kb = (ln >> 4) << 3;
  const uint32_t bt_base = (uint32_t)(uintptr_t)(&Bt[0][0]);

  auto issue_tile = [&](int iter, int buf) {
    const __bf16* src = Wt + (size_t)iter * 4096 + tid * 8;
    const uint32_t l0 = bt_base + buf * 8192 + tid * 16;
    async_copy_b128(l0, src);
    async_copy_b128(l0 + 4096, src + 2048);
  };

  const int rowA0 = b0 + (ln & 15);
  const int rowA1 = rowA0 + 16;
  v8f acc0 = {}, acc1 = {};

  issue_tile(iter0, 0);
  for (int it = 0; it < NITER / 4; ++it) {
    const int iter = iter0 + it;
    const int f = iter >> 4;
    const int h0 = (iter & 15) << 5;
    const int buf = it & 1;
    wait_async0();
    __syncthreads();  // tile `buf` fully in LDS for all waves
    if (it + 1 < NITER / 4) issue_tile(iter + 1, buf ^ 1);

    // A fragments: a[row][hl] = date[row][f] * wdiff[row][h0+hl]
    Frag fa0, fa1;
    {
      const __bf16* wd0 = wdiff_bf + (size_t)rowA0 * HID_ + h0 + kb;
      const __bf16* wd1 = wdiff_bf + (size_t)rowA1 * HID_ + h0 + kb;
      const float d0 = (float)date_bf[(size_t)rowA0 * HID_ + f];
      const float d1 = (float)date_bf[(size_t)rowA1 * HID_ + f];
      Frag t0, t1;
      t0.q[0] = *(const uint4*)wd0; t0.q[1] = *(const uint4*)(wd0 + 16);
      t1.q[0] = *(const uint4*)wd1; t1.q[1] = *(const uint4*)(wd1 + 16);
      for (int j = 0; j < 16; ++j) {
        fa0.e[j] = (__bf16)((float)t0.e[j] * d0);
        fa1.e[j] = (__bf16)((float)t1.e[j] * d1);
      }
    }
    Frag fb;
    {
      const __bf16* p = &Bt[buf][(w * 16 + (ln & 15)) * 32 + kb];
      fb.q[0] = *(const uint4*)p;
      fb.q[1] = *(const uint4*)(p + 16);
    }
    acc0 = __builtin_amdgcn_wmma_f32_16x16x32_bf16(
        false, fa0.v, false, fb.v, (short)0, acc0, false, false);
    acc1 = __builtin_amdgcn_wmma_f32_16x16x32_bf16(
        false, fa1.v, false, fb.v, (short)0, acc1, false, false);
  }

  float* dst = partial + ((size_t)ks * BSZ + b0) * NC;
  const int n = w * 16 + (ln & 15);
  const int mo = (ln >> 4) << 3;
  for (int j = 0; j < 8; ++j) {
    dst[(size_t)(mo + j) * NC + n] = acc0[j];
    dst[(size_t)(16 + mo + j) * NC + n] = acc1[j];
  }
}

// ---------------------------------------------------------------------------
// Kernel 4: out += sum over 4 K-split partials (deterministic reduction).
// ---------------------------------------------------------------------------
__global__ __launch_bounds__(256) void reduce_kernel(
    const float* __restrict__ partial, float* __restrict__ out) {
  const int i = blockIdx.x * 256 + threadIdx.x;  // 0 .. B*C-1
  float s = out[i];
  for (int ks = 0; ks < 4; ++ks) s += partial[(size_t)ks * (BSZ * NC) + i];
  out[i] = s;
}

// ---------------------------------------------------------------------------
extern "C" void kernel_launch(void* const* d_in, const int* in_sizes, int n_in,
                              void* d_out, int out_size, void* d_ws,
                              size_t ws_size, hipStream_t stream) {
  const int* idx           = (const int*)d_in[0];
  const float* date        = (const float*)d_in[1];
  const float* train_dates = (const float*)d_in[2];
  const float* mem         = (const float*)d_in[3];
  const int* train_nns     = (const int*)d_in[4];
  const float* pos_w       = (const float*)d_in[5];
  const float* pos_b       = (const float*)d_in[6];
  const float* field_w     = (const float*)d_in[7];
  const float* field_b     = (const float*)d_in[8];
  const float* adapt_w     = (const float*)d_in[9];
  const float* adapt_b     = (const float*)d_in[10];
  float* out = (float*)d_out;

  // Workspace layout (~109 MB total)
  char* ws = (char*)d_ws;
  size_t off = 0;
  float* pos_all = (float*)(ws + off);  off += (size_t)MROWS * HID_ * 4;  // 37.75MB
  __bf16* date_bf  = (__bf16*)(ws + off); off += (size_t)BSZ * HID_ * 2; // 2MB
  __bf16* wdiff_bf = (__bf16*)(ws + off); off += (size_t)BSZ * HID_ * 2; // 2MB
  float* partial = (float*)(ws + off);  off += (size_t)4 * BSZ * NC * 4; // 4MB
  __bf16* Wt     = (__bf16*)(ws + off); off += (size_t)NITER * 4096 * 2; // 64MB
  __bf16* pos_wt = (__bf16*)(ws + off);                                  // 0.5MB

  wprep_kernel<<<NITER + 8, 256, 0, stream>>>(field_w, pos_w, Wt, pos_wt);
  pos_gemm_kernel<<<MROWS / 32, 256, 0, stream>>>(
      idx, date, train_dates, train_nns, pos_wt, pos_b, pos_all);
  dist_kernel<<<BSZ, 128, 0, stream>>>(
      idx, date, mem, train_nns, adapt_w, adapt_b, field_b, pos_all,
      date_bf, wdiff_bf, out);
  field_gemm_kernel<<<256, 256, 0, stream>>>(Wt, date_bf, wdiff_bf, partial);
  reduce_kernel<<<(BSZ * NC) / 256, 256, 0, stream>>>(partial, out);
}
